// UniversalTemporalGNN_40578851012820
// MI455X (gfx1250) — compile-verified, matrix operand order checked by reading, not run on hardware
//
#include <hip/hip_runtime.h>
#include <hip/hip_bf16.h>
#include <math.h>

// ---------------------------------------------------------------------------
// UniversalTemporalGNN for MI455X (gfx1250, wave32, WMMA)
//   Heavy GEMMs (GATv2 projections)  -> v_wmma_f32_16x16x32_bf16,
//     register-tiled 16x64 per wave: A fragment reused across 4 B fragments
//     (10 b128 loads per 4 WMMAs instead of 16).
//   Edge softmax (segment max/sum)   -> f32 VALU + L2-resident atomics
//   Graphs processed in chunks of 16 so intermediates (~90MB) fit the 192MB L2
// ---------------------------------------------------------------------------

#define NH     4
#define DDIM   96
#define HDIM   384          // NH*DDIM
#define NNODE  1024
#define NEDGE  32768
#define EETOT  (NEDGE + NNODE)   // edges + self loops
#define NGRAPH 96                // B*T
#define CHUNK  16
#define NCHUNK (NGRAPH / CHUNK)
#define BB     8
#define TT     12
#define LH     96           // LSTM hidden
#define NEG_SLOPE 0.2f
#define LN_EPS    1e-5f

typedef __bf16 bf16_t;
typedef bf16_t v16bf __attribute__((ext_vector_type(16)));
typedef float  v8f   __attribute__((ext_vector_type(8)));

__device__ __forceinline__ unsigned short f2bf(float f) {
    unsigned int u = __float_as_uint(f);
    u += 0x7FFFu + ((u >> 16) & 1u);        // round-to-nearest-even
    return (unsigned short)(u >> 16);
}

__device__ __forceinline__ void atomicMaxF(float* a, float v) {
    unsigned int* ai = (unsigned int*)a;
    unsigned int old = *ai;
    while (v > __uint_as_float(old)) {
        unsigned int assumed = old;
        old = atomicCAS(ai, assumed, __float_as_uint(v));
        if (old == assumed) break;
    }
}

// ------------------------------ utility ------------------------------------
__global__ void fill_f32(float* p, float v, int n) {
    for (int i = blockIdx.x * blockDim.x + threadIdx.x; i < n; i += gridDim.x * blockDim.x)
        p[i] = v;
}

// W (fin x 384) f32  ->  WT (384 x Kp) bf16, zero padded in K
__global__ void conv_wT(const float* __restrict__ W, unsigned short* __restrict__ WT,
                        int fin, int Kp) {
    int idx = blockIdx.x * blockDim.x + threadIdx.x;
    if (idx >= HDIM * Kp) return;
    int n = idx / Kp, k = idx - n * Kp;
    WT[idx] = (k < fin) ? f2bf(W[(size_t)k * HDIM + n]) : (unsigned short)0;
}

// x_sequence (96,1024,40) f32 -> chunk bf16 (CHUNK,1024,64) zero padded
__global__ void conv_x0(const float* __restrict__ x, unsigned short* __restrict__ xb,
                        int gStart) {
    int idx = blockIdx.x * blockDim.x + threadIdx.x;
    if (idx >= CHUNK * NNODE * 64) return;
    int g = idx / (NNODE * 64);
    int rem = idx - g * NNODE * 64;
    int n = rem / 64, k = rem - n * 64;
    xb[idx] = (k < 40) ? f2bf(x[((size_t)(gStart + g) * NNODE + n) * 40 + k])
                       : (unsigned short)0;
}

// ------------------------------ WMMA GEMM ----------------------------------
// C(MxN) = A(MxK,bf16,row-major) * B(KxN) + bias;  BT is B transposed (NxK bf16)
// One 16x64 strip per wave (4 consecutive 16x16 N-tiles): the A fragment is
// loaded once per K-step and reused for 4 WMMAs. K multiple of 32, N of 64.
// Fragment per ISA 16-bit layout: lane L holds row M=L%16, K chunks
// [kh*8..kh*8+7] and [16+kh*8..+7] with kh = L/16 -> two b128 loads each.
__global__ void wmma_gemm_bias(const unsigned short* __restrict__ A,
                               const unsigned short* __restrict__ BT,
                               const float* __restrict__ bias,
                               float* __restrict__ C, int M, int N, int K) {
    int wave = threadIdx.x >> 5;
    int lane = threadIdx.x & 31;
    int tilesG = N >> 6;                             // groups of 4 N-tiles
    int tile = blockIdx.x * (blockDim.x >> 5) + wave;
    if (tile >= (M >> 4) * tilesG) return;          // wave-uniform: EXEC stays full
    int tm = tile / tilesG, tg = tile - tm * tilesG;

    int mrow  = (tm << 4) + (lane & 15);
    int ncol0 = (tg << 6) + (lane & 15);             // N-tile j adds 16*j
    int khalf = (lane >> 4) << 3;                    // 0 or 8

    const unsigned short* arow = A  + (size_t)mrow * K;
    const unsigned short* brow = BT + (size_t)ncol0 * K;

    v8f acc[4];
#pragma unroll
    for (int j = 0; j < 4; ++j) acc[j] = (v8f){0.f,0.f,0.f,0.f,0.f,0.f,0.f,0.f};

    for (int k0 = 0; k0 < K; k0 += 32) {
        union { v16bf v; uint4 u[2]; } a, b[4];
        a.u[0] = *(const uint4*)(arow + k0 + khalf);
        a.u[1] = *(const uint4*)(arow + k0 + 16 + khalf);
#pragma unroll
        for (int j = 0; j < 4; ++j) {
            const unsigned short* bj = brow + (size_t)(16 * j) * K;
            b[j].u[0] = *(const uint4*)(bj + k0 + khalf);
            b[j].u[1] = *(const uint4*)(bj + k0 + 16 + khalf);
        }
#pragma unroll
        for (int j = 0; j < 4; ++j)
            acc[j] = __builtin_amdgcn_wmma_f32_16x16x32_bf16(
                         false, a.v, false, b[j].v, (short)0, acc[j], false, false);
    }

    int mb = (tm << 4) + ((lane >> 4) << 3);         // C/D layout: r -> row mb+r
#pragma unroll
    for (int j = 0; j < 4; ++j) {
        int ncol = ncol0 + 16 * j;
        float bv = bias[ncol];
#pragma unroll
        for (int r = 0; r < 8; ++r)
            C[(size_t)(mb + r) * N + ncol] = acc[j][r] + bv;
    }
}

// ------------------------------ GATv2 edge passes --------------------------
// idx -> (g, e, h);  e >= NEDGE are self loops.
__global__ void gat_pass1(const float* __restrict__ xl, const float* __restrict__ xr,
                          const int* __restrict__ ei, const float* __restrict__ att,
                          float* __restrict__ logitb, float* __restrict__ maxb) {
    int idx = blockIdx.x * blockDim.x + threadIdx.x;
    if (idx >= CHUNK * EETOT * NH) return;
    int g = idx / (EETOT * NH);
    int rem = idx - g * EETOT * NH;
    int e = rem / NH, h = rem - e * NH;
    int src = (e < NEDGE) ? ei[e]         : (e - NEDGE);
    int dst = (e < NEDGE) ? ei[NEDGE + e] : (e - NEDGE);
    const float* pl = xl + ((size_t)(g * NNODE + src)) * HDIM + h * DDIM;
    const float* pr = xr + ((size_t)(g * NNODE + dst)) * HDIM + h * DDIM;
    const float* pa = att + h * DDIM;
    float acc = 0.f;
#pragma unroll 4
    for (int d = 0; d < DDIM; ++d) {
        float v = pl[d] + pr[d];
        v = (v > 0.f) ? v : NEG_SLOPE * v;
        acc = fmaf(pa[d], v, acc);
    }
    logitb[idx] = acc;
    atomicMaxF(&maxb[(g * NNODE + dst) * NH + h], acc);
}

__global__ void gat_pass2(const int* __restrict__ ei, const float* __restrict__ logitb,
                          const float* __restrict__ maxb, float* __restrict__ denom) {
    int idx = blockIdx.x * blockDim.x + threadIdx.x;
    if (idx >= CHUNK * EETOT * NH) return;
    int g = idx / (EETOT * NH);
    int rem = idx - g * EETOT * NH;
    int e = rem / NH, h = rem - e * NH;
    int dst = (e < NEDGE) ? ei[NEDGE + e] : (e - NEDGE);
    float ex = __expf(logitb[idx] - maxb[(g * NNODE + dst) * NH + h]);
    atomicAdd(&denom[(g * NNODE + dst) * NH + h], ex);
}

__global__ void gat_pass3(const float* __restrict__ xl, const int* __restrict__ ei,
                          const float* __restrict__ logitb, const float* __restrict__ maxb,
                          const float* __restrict__ denom, float* __restrict__ agg) {
    int idx = blockIdx.x * blockDim.x + threadIdx.x;
    if (idx >= CHUNK * EETOT * NH) return;
    int g = idx / (EETOT * NH);
    int rem = idx - g * EETOT * NH;
    int e = rem / NH, h = rem - e * NH;
    int src = (e < NEDGE) ? ei[e]         : (e - NEDGE);
    int dst = (e < NEDGE) ? ei[NEDGE + e] : (e - NEDGE);
    int seg = (g * NNODE + dst) * NH + h;
    float alpha = __expf(logitb[idx] - maxb[seg]) / (denom[seg] + 1e-16f);
    const float* pl = xl + ((size_t)(g * NNODE + src)) * HDIM + h * DDIM;
    float* po = agg + ((size_t)(g * NNODE + dst)) * HDIM + h * DDIM;
#pragma unroll 4
    for (int d = 0; d < DDIM; ++d)
        atomicAdd(&po[d], alpha * pl[d]);
}

// bias + LayerNorm + ELU over 384 features; one wave per node (wave32)
__global__ void gat_ln_elu(const float* __restrict__ agg, const float* __restrict__ gbias,
                           const float* __restrict__ gamma, const float* __restrict__ beta,
                           unsigned short* __restrict__ xb_out, float* __restrict__ f32_out) {
    int node = blockIdx.x * 8 + (threadIdx.x >> 5);
    int lane = threadIdx.x & 31;
    if (node >= CHUNK * NNODE) return;
    const float* row = agg + (size_t)node * HDIM;
    float v[12];
    float s = 0.f;
#pragma unroll
    for (int i = 0; i < 12; ++i) {
        int f = lane + i * 32;
        v[i] = row[f] + gbias[f];
        s += v[i];
    }
    for (int off = 16; off; off >>= 1) s += __shfl_xor(s, off, 32);
    float mu = s * (1.f / HDIM);
    float q = 0.f;
#pragma unroll
    for (int i = 0; i < 12; ++i) { float d = v[i] - mu; q += d * d; }
    for (int off = 16; off; off >>= 1) q += __shfl_xor(q, off, 32);
    float rstd = rsqrtf(q * (1.f / HDIM) + LN_EPS);
#pragma unroll
    for (int i = 0; i < 12; ++i) {
        int f = lane + i * 32;
        float y = (v[i] - mu) * rstd * gamma[f] + beta[f];
        y = (y > 0.f) ? y : (__expf(y) - 1.f);      // ELU
        xb_out[(size_t)node * HDIM + f] = f2bf(y);
        if (f32_out) f32_out[(size_t)node * HDIM + f] = y;
    }
}

// mean over 1024 nodes -> node_emb[(gStart+g)*384 + f]
__global__ void node_mean(const float* __restrict__ xf, float* __restrict__ nemb, int gStart) {
    int idx = blockIdx.x * blockDim.x + threadIdx.x;
    if (idx >= CHUNK * HDIM) return;
    int g = idx / HDIM, f = idx - g * HDIM;
    const float* p = xf + (size_t)g * NNODE * HDIM + f;
    float s = 0.f;
    for (int n = 0; n < NNODE; ++n) s += p[(size_t)n * HDIM];
    nemb[(gStart + g) * HDIM + f] = s * (1.f / NNODE);
}

// ------------------------------ BiLSTM -------------------------------------
// grid (batch=8, dir=2), block 384 (one thread per gate row)
__global__ void lstm_bidir(const float* __restrict__ x, int inDim,
                           const float* WihF, const float* WhhF,
                           const float* bihF, const float* bhhF,
                           const float* WihR, const float* WhhR,
                           const float* bihR, const float* bhhR,
                           float* __restrict__ y) {
    int b = blockIdx.x, dir = blockIdx.y, tid = threadIdx.x;
    const float* Wih = dir ? WihR : WihF;
    const float* Whh = dir ? WhhR : WhhF;
    const float* bih = dir ? bihR : bihF;
    const float* bhh = dir ? bhhR : bhhF;
    __shared__ float hs[LH], cs[LH], gates[4 * LH];
    if (tid < LH) { hs[tid] = 0.f; cs[tid] = 0.f; }
    __syncthreads();
    for (int step = 0; step < TT; ++step) {
        int t = dir ? (TT - 1 - step) : step;
        const float* xrow = x + (size_t)(b * TT + t) * inDim;
        float acc = bih[tid] + bhh[tid];
        const float* wi = Wih + (size_t)tid * inDim;
        for (int k = 0; k < inDim; ++k) acc = fmaf(xrow[k], wi[k], acc);
        const float* wh = Whh + (size_t)tid * LH;
        for (int k = 0; k < LH; ++k) acc = fmaf(hs[k], wh[k], acc);
        gates[tid] = acc;
        __syncthreads();
        if (tid < LH) {
            float ig = 1.f / (1.f + __expf(-gates[tid]));
            float fg = 1.f / (1.f + __expf(-gates[LH + tid]));
            float gg = tanhf(gates[2 * LH + tid]);
            float og = 1.f / (1.f + __expf(-gates[3 * LH + tid]));
            float c = fg * cs[tid] + ig * gg;
            cs[tid] = c;
            float hv = og * tanhf(c);
            hs[tid] = hv;
            y[(size_t)(b * TT + t) * (2 * LH) + dir * LH + tid] = hv;
        }
        __syncthreads();
    }
}

// ------------------------------ heads --------------------------------------
__global__ void head_linear(const float* __restrict__ te, int teStride,
                            const float* __restrict__ W, const float* __restrict__ b,
                            float* __restrict__ out, int inDim, int outDim) {
    int idx = blockIdx.x * blockDim.x + threadIdx.x;
    if (idx >= BB * outDim) return;
    int bb = idx / outDim, row = idx - bb * outDim;
    const float* t = te + (size_t)bb * teStride;
    const float* w = W + (size_t)row * inDim;
    float acc = b[row];
    for (int k = 0; k < inDim; ++k) acc = fmaf(t[k], w[k], acc);
    out[(size_t)bb * outDim + row] = acc;
}

// z = LN(relu(te@W1^T+b1)); one block per batch, 96 threads
__global__ void rul_z(const float* __restrict__ te, int teStride,
                      const float* __restrict__ W1, const float* __restrict__ b1,
                      const float* __restrict__ gamma, const float* __restrict__ beta,
                      float* __restrict__ zbuf) {
    int b = blockIdx.x, j = threadIdx.x;
    const float* t = te + (size_t)b * teStride;
    const float* w = W1 + (size_t)j * (2 * LH);
    float acc = b1[j];
    for (int k = 0; k < 2 * LH; ++k) acc = fmaf(t[k], w[k], acc);
    acc = fmaxf(acc, 0.f);
    __shared__ float z[LH];
    __shared__ float mu_s, rstd_s;
    z[j] = acc;
    __syncthreads();
    if (j == 0) {
        float s = 0.f;
        for (int k = 0; k < LH; ++k) s += z[k];
        float mu = s * (1.f / LH);
        float q = 0.f;
        for (int k = 0; k < LH; ++k) { float d = z[k] - mu; q += d * d; }
        mu_s = mu;
        rstd_s = rsqrtf(q * (1.f / LH) + LN_EPS);
    }
    __syncthreads();
    zbuf[(size_t)b * LH + j] = (acc - mu_s) * rstd_s * gamma[j] + beta[j];
}

__global__ void rul_out(const float* __restrict__ zbuf, const float* __restrict__ W2,
                        const float* __restrict__ b2, float* __restrict__ out) {
    int idx = blockIdx.x * blockDim.x + threadIdx.x;
    if (idx >= BB * NNODE) return;
    int b = idx / NNODE, row = idx - b * NNODE;
    const float* z = zbuf + (size_t)b * LH;
    const float* w = W2 + (size_t)row * LH;
    float acc = b2[row];
    for (int k = 0; k < LH; ++k) acc = fmaf(z[k], w[k], acc);
    out[(size_t)b * NNODE + row] = 1.f / (1.f + __expf(-acc));
}

// ---------------------------------------------------------------------------
extern "C" void kernel_launch(void* const* d_in, const int* in_sizes, int n_in,
                              void* d_out, int out_size, void* d_ws, size_t ws_size,
                              hipStream_t stream) {
    // Input order = setup_inputs() dict insertion order, nested containers in
    // written order:
    //  0 x_sequence
    //  1+6i: gat[i] {Wl, bl, Wr, br, att, bias}
    //  19..24: ln (g0,b0,g1,b1,g2,b2)
    //  25+8j: lstm[j] {fwd:(Wih,Whh,bih,bhh), bwd:(Wih,Whh,bih,bhh)}
    //  41,42: anomaly {W,b}
    //  43+6k: rul['5','15','30'] {W1,b1,g,beta,W2,b2}
    //  61: edge_index (2,E) int32
    const float* x_seq = (const float*)d_in[0];
    const float *Wl[3], *blp[3], *Wr[3], *brp[3], *attp[3], *gbias[3];
    for (int i = 0; i < 3; ++i) {
        int b0 = 1 + i * 6;
        Wl[i]    = (const float*)d_in[b0 + 0];
        blp[i]   = (const float*)d_in[b0 + 1];
        Wr[i]    = (const float*)d_in[b0 + 2];
        brp[i]   = (const float*)d_in[b0 + 3];
        attp[i]  = (const float*)d_in[b0 + 4];
        gbias[i] = (const float*)d_in[b0 + 5];
    }
    const float *lng[3], *lnb[3];
    for (int i = 0; i < 3; ++i) {
        lng[i] = (const float*)d_in[19 + 2 * i];
        lnb[i] = (const float*)d_in[20 + 2 * i];
    }
    const float* lstm[2][8];   // [layer][fwd Wih,Whh,bih,bhh, bwd Wih,Whh,bih,bhh]
    for (int j = 0; j < 2; ++j)
        for (int k = 0; k < 8; ++k)
            lstm[j][k] = (const float*)d_in[25 + 8 * j + k];
    const float* anomW = (const float*)d_in[41];
    const float* anomB = (const float*)d_in[42];
    const float *rW1[3], *rb1[3], *rg[3], *rbeta[3], *rW2[3], *rb2[3];
    for (int k = 0; k < 3; ++k) {     // order 5, 15, 30 (input and output)
        int b0 = 43 + k * 6;
        rW1[k]   = (const float*)d_in[b0 + 0];
        rb1[k]   = (const float*)d_in[b0 + 1];
        rg[k]    = (const float*)d_in[b0 + 2];
        rbeta[k] = (const float*)d_in[b0 + 3];
        rW2[k]   = (const float*)d_in[b0 + 4];
        rb2[k]   = (const float*)d_in[b0 + 5];
    }
    const int* ei = (const int*)d_in[61];
    float* out = (float*)d_out;

    // ---- workspace carve (aligned 256B) ----
    char* ws = (char*)d_ws;
    size_t off = 0;
    auto alloc = [&](size_t bytes) -> void* {
        void* p = ws + off;
        off = (off + bytes + 255) & ~(size_t)255;
        return p;
    };
    unsigned short *WlT[3], *WrT[3];
    for (int i = 0; i < 3; ++i) {
        WlT[i] = (unsigned short*)alloc((size_t)HDIM * HDIM * 2);
        WrT[i] = (unsigned short*)alloc((size_t)HDIM * HDIM * 2);
    }
    unsigned short* xb = (unsigned short*)alloc((size_t)CHUNK * NNODE * HDIM * 2);
    float* xlb  = (float*)alloc((size_t)CHUNK * NNODE * HDIM * 4);
    float* xrb  = (float*)alloc((size_t)CHUNK * NNODE * HDIM * 4);
    float* aggb = (float*)alloc((size_t)CHUNK * NNODE * HDIM * 4);
    float* maxb = (float*)alloc((size_t)CHUNK * NNODE * NH * 4);
    float* denb = (float*)alloc((size_t)CHUNK * NNODE * NH * 4);
    float* logb = (float*)alloc((size_t)CHUNK * EETOT * NH * 4);
    float* nemb = (float*)alloc((size_t)NGRAPH * HDIM * 4);
    float* y1   = (float*)alloc((size_t)BB * TT * 2 * LH * 4);
    float* y2   = (float*)alloc((size_t)BB * TT * 2 * LH * 4);
    float* zb   = (float*)alloc((size_t)3 * BB * LH * 4);

    // ---- weight convert+transpose (once) ----
    const int fins[3] = {40, HDIM, HDIM};
    const int Kps[3]  = {64, HDIM, HDIM};
    for (int i = 0; i < 3; ++i) {
        int tot = HDIM * Kps[i];
        conv_wT<<<(tot + 255) / 256, 256, 0, stream>>>(Wl[i], WlT[i], fins[i], Kps[i]);
        conv_wT<<<(tot + 255) / 256, 256, 0, stream>>>(Wr[i], WrT[i], fins[i], Kps[i]);
    }

    // ---- GAT stack, chunked over graphs ----
    const int M = CHUNK * NNODE;
    const int etot = CHUNK * EETOT * NH;
    for (int c = 0; c < NCHUNK; ++c) {
        int gStart = c * CHUNK;
        for (int layer = 0; layer < 3; ++layer) {
            int Kp = Kps[layer];
            if (layer == 0) {
                int tot = CHUNK * NNODE * 64;
                conv_x0<<<(tot + 255) / 256, 256, 0, stream>>>(x_seq, xb, gStart);
            }
            int tiles = (M / 16) * (HDIM / 64);      // 16x64 strip per wave
            wmma_gemm_bias<<<(tiles + 7) / 8, 256, 0, stream>>>(xb, WlT[layer], blp[layer],
                                                                xlb, M, HDIM, Kp);
            wmma_gemm_bias<<<(tiles + 7) / 8, 256, 0, stream>>>(xb, WrT[layer], brp[layer],
                                                                xrb, M, HDIM, Kp);
            fill_f32<<<256, 256, 0, stream>>>(maxb, -3.0e38f, CHUNK * NNODE * NH);
            fill_f32<<<256, 256, 0, stream>>>(denb, 0.f, CHUNK * NNODE * NH);
            fill_f32<<<2048, 256, 0, stream>>>(aggb, 0.f, CHUNK * NNODE * HDIM);
            gat_pass1<<<(etot + 255) / 256, 256, 0, stream>>>(xlb, xrb, ei, attp[layer],
                                                              logb, maxb);
            gat_pass2<<<(etot + 255) / 256, 256, 0, stream>>>(ei, logb, maxb, denb);
            gat_pass3<<<(etot + 255) / 256, 256, 0, stream>>>(xlb, ei, logb, maxb, denb, aggb);
            gat_ln_elu<<<(CHUNK * NNODE + 7) / 8, 256, 0, stream>>>(
                aggb, gbias[layer], lng[layer], lnb[layer], xb,
                (layer == 2) ? xlb : (float*)nullptr);
        }
        int tot = CHUNK * HDIM;
        node_mean<<<(tot + 255) / 256, 256, 0, stream>>>(xlb, nemb, gStart);
    }

    // ---- BiLSTM stack ----
    lstm_bidir<<<dim3(BB, 2), 4 * LH, 0, stream>>>(nemb, HDIM,
        lstm[0][0], lstm[0][1], lstm[0][2], lstm[0][3],
        lstm[0][4], lstm[0][5], lstm[0][6], lstm[0][7], y1);
    lstm_bidir<<<dim3(BB, 2), 4 * LH, 0, stream>>>(y1, 2 * LH,
        lstm[1][0], lstm[1][1], lstm[1][2], lstm[1][3],
        lstm[1][4], lstm[1][5], lstm[1][6], lstm[1][7], y2);

    // ---- heads; te = y2[:, T-1, :], per-batch stride T*2L ----
    const float* te = y2 + (size_t)(TT - 1) * 2 * LH;
    head_linear<<<(BB * NNODE + 255) / 256, 256, 0, stream>>>(te, TT * 2 * LH, anomW, anomB,
                                                              out, 2 * LH, NNODE);
    for (int k = 0; k < 3; ++k) {     // output slots: rul5, rul15, rul30
        rul_z<<<BB, LH, 0, stream>>>(te, TT * 2 * LH, rW1[k], rb1[k], rg[k], rbeta[k],
                                     zb + (size_t)k * BB * LH);
        rul_out<<<(BB * NNODE + 255) / 256, 256, 0, stream>>>(zb + (size_t)k * BB * LH,
                                                              rW2[k], rb2[k],
                                                              out + (size_t)(1 + k) * BB * NNODE);
    }
}
